// MoeKANLayer_71322226917975
// MI455X (gfx1250) — compile-verified, matrix operand order checked by reading, not run on hardware
//
#include <hip/hip_runtime.h>

// ---------------- types ----------------
typedef __attribute__((ext_vector_type(16))) __bf16       v16bf;
typedef __attribute__((ext_vector_type(8)))  float        v8f;
typedef __attribute__((ext_vector_type(4)))  float        v4f;
typedef __attribute__((ext_vector_type(4)))  unsigned int v4u;
typedef __attribute__((ext_vector_type(4)))  int          v4i;

union Frag  { v16bf v; v4u q[2]; };

// ---------------- constants ----------------
#define NTOK   2048
#define HID    512
#define DFF    1024
#define NEXP   8
#define K1     4608       // HID + HID*8
#define N12    2048       // DFF*2 (h1|h2)
#define K3     9216       // DFF + DFF*8
#define OUTD   512
#define LDSP   40         // LDS row pitch in bf16 (80 B, conflict-spread)

// ---------------- optional CDNA5 async global->LDS path ----------------
#if defined(__has_builtin)
#if __has_builtin(__builtin_amdgcn_global_load_async_to_lds_b128) && \
    __has_builtin(__builtin_amdgcn_s_wait_asynccnt)
#define HAVE_ASYNC_LDS 1
#endif
#endif

#ifdef HAVE_ASYNC_LDS
__device__ __forceinline__ void async_cp16(const unsigned short* g, unsigned short* l) {
    __builtin_amdgcn_global_load_async_to_lds_b128((v4i*)g, (v4i*)l, 0, 0);
}
#endif

// ---------------- bf16 conversion (HW packed cvt when available) ----------------
#if defined(__has_builtin)
#if __has_builtin(__builtin_amdgcn_cvt_pk_bf16_f32)
#define HAVE_PK_BF16 1
#endif
#endif

#ifdef HAVE_PK_BF16
typedef __attribute__((ext_vector_type(2))) __bf16 v2bf;
__device__ __forceinline__ unsigned int pkbf(float a, float b) {
    v2bf r = __builtin_amdgcn_cvt_pk_bf16_f32(a, b);
    return __builtin_bit_cast(unsigned int, r);
}
#else
__device__ __forceinline__ unsigned int pkbf(float a, float b) {
    unsigned int ua = __float_as_uint(a);
    ua += 0x7FFFu + ((ua >> 16) & 1u);
    unsigned int ub = __float_as_uint(b);
    ub += 0x7FFFu + ((ub >> 16) & 1u);
    return (ua >> 16) | (ub & 0xFFFF0000u);
}
#endif

__device__ __forceinline__ unsigned short f2bf(float f) {
    return (unsigned short)(pkbf(f, 0.0f) & 0xFFFFu);
}
__device__ __forceinline__ float silu_f(float v) {
    return v * (1.0f / (1.0f + __expf(-v)));
}

// Cubic B-spline bases on uniform KAN grid: knots t_j = (j-3)*0.4 - 1, j=0..11
__device__ __forceinline__ void bspline8(float x, float* out8) {
    const float h = 2.0f / 5.0f;
    float t[12];
#pragma unroll
    for (int j = 0; j < 12; ++j) t[j] = (float)(j - 3) * h - 1.0f;
    float b[11];
#pragma unroll
    for (int j = 0; j < 11; ++j) b[j] = (x >= t[j] && x < t[j + 1]) ? 1.0f : 0.0f;
#pragma unroll
    for (int ord = 1; ord <= 3; ++ord) {
#pragma unroll
        for (int j = 0; j < 10; ++j) {
            if (j < 11 - ord) {
                float left  = (x - t[j])           / (t[j + ord]     - t[j]);
                float right = (t[j + ord + 1] - x) / (t[j + ord + 1] - t[j + 1]);
                b[j] = left * b[j] + right * b[j + 1];
            }
        }
    }
#pragma unroll
    for (int c = 0; c < 8; ++c) out8[c] = b[c];
}

// ---------------- small kernels ----------------
__global__ void zero_f32(float* __restrict__ p, int n) {
    int i = blockIdx.x * blockDim.x + threadIdx.x;
    if (i < n) p[i] = 0.0f;
}

// One wave per token: 8 gate logits, top-2 softmax -> dense (N, E) gate weights
__global__ void gate_kernel(const float* __restrict__ x, const float* __restrict__ gw,
                            float* __restrict__ gates) {
    int n = blockIdx.x;
    int lane = threadIdx.x;
    float acc[NEXP];
#pragma unroll
    for (int e = 0; e < NEXP; ++e) acc[e] = 0.0f;
    for (int k = lane; k < HID; k += 32) {
        float xv = x[(size_t)n * HID + k];
#pragma unroll
        for (int e = 0; e < NEXP; ++e) acc[e] += xv * gw[(size_t)e * HID + k];
    }
#pragma unroll
    for (int off = 16; off > 0; off >>= 1) {
#pragma unroll
        for (int e = 0; e < NEXP; ++e) acc[e] += __shfl_down(acc[e], off, 32);
    }
    if (lane == 0) {
        int i0 = 0;
#pragma unroll
        for (int e = 1; e < NEXP; ++e) if (acc[e] > acc[i0]) i0 = e;
        int i1 = (i0 == 0) ? 1 : 0;
#pragma unroll
        for (int e = 0; e < NEXP; ++e) if (e != i0 && acc[e] > acc[i1]) i1 = e;
        float m  = fmaxf(acc[i0], acc[i1]);
        float e0 = __expf(acc[i0] - m), e1 = __expf(acc[i1] - m);
        float s  = e0 + e1;
#pragma unroll
        for (int e = 0; e < NEXP; ++e) gates[(size_t)n * NEXP + e] = 0.0f;
        gates[(size_t)n * NEXP + i0] = e0 / s;
        gates[(size_t)n * NEXP + i1] = e1 / s;
    }
}

// A1 = [silu(x) | bsplines(x)] bf16, (NTOK, K1)
__global__ void prep_a1(const float* __restrict__ x, unsigned short* __restrict__ A1) {
    int tid = blockIdx.x * blockDim.x + threadIdx.x;
    if (tid >= NTOK * HID) return;
    int n = tid >> 9, i = tid & (HID - 1);
    float v = x[tid];
    A1[(size_t)n * K1 + i] = f2bf(silu_f(v));
    float b[8];
    bspline8(v, b);
    v4u q;
    q[0] = pkbf(b[0], b[1]); q[1] = pkbf(b[2], b[3]);
    q[2] = pkbf(b[4], b[5]); q[3] = pkbf(b[6], b[7]);
    *(v4u*)&A1[(size_t)n * K1 + HID + (size_t)i * 8] = q;
}

// h = h1*h2 ; A3 = [silu(h) | bsplines(h)] bf16, (NTOK, K3)
__global__ void act_kernel(const float* __restrict__ H12, unsigned short* __restrict__ A3) {
    int tid = blockIdx.x * blockDim.x + threadIdx.x;
    if (tid >= NTOK * DFF) return;
    int n = tid >> 10, j = tid & (DFF - 1);
    float h = H12[(size_t)n * N12 + j] * H12[(size_t)n * N12 + DFF + j];
    A3[(size_t)n * K3 + j] = f2bf(silu_f(h));
    float b[8];
    bspline8(h, b);
    v4u q;
    q[0] = pkbf(b[0], b[1]); q[1] = pkbf(b[2], b[3]);
    q[2] = pkbf(b[4], b[5]); q[3] = pkbf(b[6], b[7]);
    *(v4u*)&A3[(size_t)n * K3 + DFF + (size_t)j * 8] = q;
}

// ---------------- compile-time GEMM configs ----------------
template <int MODE> struct GemmCfg;
template <> struct GemmCfg<0> { static constexpr int K = K1, LDC = N12, F = HID; };
template <> struct GemmCfg<1> { static constexpr int K = K3, LDC = OUTD, F = DFF; };

// fused KAN-weight fetch: fp32 -> scaled bf16 (16-wide K chunk)
// K layout: [0, F) = base weights ; [F, 9F) = spline (i,c) with scaler[i].
template <int MODE>
__device__ __forceinline__ void fetchB(const float* __restrict__ bw,
                                       const float* __restrict__ sw,
                                       const float* __restrict__ cw,
                                       int oo, int kk, v4u& o0, v4u& o1) {
    constexpr int F = GemmCfg<MODE>::F;
    v4f f0, f1, f2, f3;
    float c0 = 1.0f, c1 = 1.0f;
    if (kk < F) {
        const v4f* p = (const v4f*)(bw + (size_t)oo * F + kk);
        f0 = p[0]; f1 = p[1]; f2 = p[2]; f3 = p[3];
    } else {
        int i0 = (kk - F) >> 3;                       // even (kk is 16-aligned)
        const v4f* p = (const v4f*)(sw + ((size_t)oo * F + i0) * 8);
        f0 = p[0]; f1 = p[1]; f2 = p[2]; f3 = p[3];
        c0 = cw[(size_t)oo * F + i0];
        c1 = cw[(size_t)oo * F + i0 + 1];
    }
    o0[0] = pkbf(f0[0] * c0, f0[1] * c0); o0[1] = pkbf(f0[2] * c0, f0[3] * c0);
    o0[2] = pkbf(f1[0] * c0, f1[1] * c0); o0[3] = pkbf(f1[2] * c0, f1[3] * c0);
    o1[0] = pkbf(f2[0] * c1, f2[1] * c1); o1[1] = pkbf(f2[2] * c1, f2[3] * c1);
    o1[2] = pkbf(f3[0] * c1, f3[1] * c1); o1[3] = pkbf(f3[2] * c1, f3[3] * c1);
}

__device__ __forceinline__ void ldfrag(Frag& f, const unsigned short* s, int row, int half) {
    f.q[0] = *(const v4u*)&s[row * LDSP + half * 8];
    f.q[1] = *(const v4u*)&s[row * LDSP + 16 + half * 8];
}

// ---------------- WMMA GEMM with fused weight conversion ----------------
// C(M,N) = A(M,K) * Wcat(N,K)^T.  256 threads = 8 waves; 128x128 block tile;
// each wave 32x64 (2x4 of 16x16 WMMA); double-buffered LDS, 1 barrier/step.
// A-tile staged via GLOBAL_LOAD_ASYNC_TO_LDS when available.
// MODE 0: N covers [layer1 | layer2] rows, C = H12. MODE 1: gated += into out.
template <int MODE>
__global__ __launch_bounds__(256)
void gemm_kan(const unsigned short* __restrict__ A,
              const float* __restrict__ bw1, const float* __restrict__ sw1,
              const float* __restrict__ cw1, const float* __restrict__ bw2,
              const float* __restrict__ sw2, const float* __restrict__ cw2,
              float* __restrict__ C, const float* __restrict__ gates, int expert) {
    constexpr int K   = GemmCfg<MODE>::K;
    constexpr int LDC = GemmCfg<MODE>::LDC;

    __shared__ unsigned short sA[2][128 * LDSP];
    __shared__ unsigned short sB[2][128 * LDSP];

    const int tileN = blockIdx.x * 128;
    const int tileM = blockIdx.y * 128;
    const int tid   = threadIdx.x;
    const int wave  = tid >> 5;
    const int lane  = tid & 31;
    const int half  = lane >> 4;
    const int idx   = lane & 15;
    const int wm    = (wave & 3) * 32;
    const int wn    = (wave >> 2) * 64;
    const int lrow  = tid >> 1;     // 0..127: staged row
    const int lseg  = tid & 1;      // 16-element K sub-chunk

    // weight-row pointers for this thread's staged B row
    const int o = tileN + lrow;
    const float* bw = bw1; const float* sw = sw1; const float* cw = cw1;
    int oo = o;
    if (MODE == 0 && o >= DFF) { bw = bw2; sw = sw2; cw = cw2; oo = o - DFF; }

    const unsigned short* Arow = A + (size_t)(tileM + lrow) * K + lseg * 16;
    const int kbase = lseg * 16;
    unsigned short* sArow0 = &sA[0][lrow * LDSP + lseg * 16];
    unsigned short* sArow1 = &sA[1][lrow * LDSP + lseg * 16];
    unsigned short* sBrow0 = &sB[0][lrow * LDSP + lseg * 16];
    unsigned short* sBrow1 = &sB[1][lrow * LDSP + lseg * 16];

    v8f acc[2][4];
#pragma unroll
    for (int i = 0; i < 2; ++i)
#pragma unroll
        for (int j = 0; j < 4; ++j) acc[i][j] = (v8f){};

    // prologue: stage K-step 0 into buffer 0
    {
#ifdef HAVE_ASYNC_LDS
        async_cp16(Arow,     sArow0);
        async_cp16(Arow + 8, sArow0 + 8);
#else
        const v4u* ga = (const v4u*)Arow;
        v4u ra0 = ga[0], ra1 = ga[1];
        v4u* la = (v4u*)sArow0;
        la[0] = ra0; la[1] = ra1;
#endif
        v4u rb0, rb1;
        fetchB<MODE>(bw, sw, cw, oo, kbase, rb0, rb1);
        v4u* lb = (v4u*)sBrow0;
        lb[0] = rb0; lb[1] = rb1;
#ifdef HAVE_ASYNC_LDS
        __builtin_amdgcn_s_wait_asynccnt(0);
#endif
    }
    __syncthreads();

    constexpr int steps = K >> 5;
    for (int s = 0; s < steps; ++s) {
        const int p = s & 1;
        const bool more = (s + 1 < steps);
        v4u nb0, nb1;
#ifndef HAVE_ASYNC_LDS
        v4u na0, na1;
#endif
        if (more) {
#ifdef HAVE_ASYNC_LDS
            unsigned short* dst = p ? sArow0 : sArow1;
            async_cp16(Arow + (s + 1) * 32,     dst);
            async_cp16(Arow + (s + 1) * 32 + 8, dst + 8);
#else
            const v4u* ga = (const v4u*)(Arow + (s + 1) * 32);
            na0 = ga[0]; na1 = ga[1];
#endif
            fetchB<MODE>(bw, sw, cw, oo, (s + 1) * 32 + kbase, nb0, nb1);
        }

        // compute on buffer p
        Frag fa[2], fb[4];
        ldfrag(fa[0], sA[p], wm + idx,      half);
        ldfrag(fa[1], sA[p], wm + 16 + idx, half);
#pragma unroll
        for (int j = 0; j < 4; ++j) ldfrag(fb[j], sB[p], wn + 16 * j + idx, half);
#pragma unroll
        for (int i = 0; i < 2; ++i)
#pragma unroll
            for (int j = 0; j < 4; ++j)
                acc[i][j] = __builtin_amdgcn_wmma_f32_16x16x32_bf16(
                    false, fa[i].v, false, fb[j].v, (short)0, acc[i][j], false, false);

        if (more) {
#ifndef HAVE_ASYNC_LDS
            v4u* la = (v4u*)(p ? sArow0 : sArow1);
            la[0] = na0; la[1] = na1;
#endif
            v4u* lb = (v4u*)(p ? sBrow0 : sBrow1);
            lb[0] = nb0; lb[1] = nb1;
#ifdef HAVE_ASYNC_LDS
            __builtin_amdgcn_s_wait_asynccnt(0);
#endif
            __syncthreads();
        }
    }

    // epilogue: constexpr LDC -> immediate-offset stores
    const int rbase = tileM + wm + half * 8;
    float* cbase = C + (size_t)rbase * LDC + (tileN + wn + idx);
    const float* gbase = (MODE == 1) ? (gates + (size_t)rbase * NEXP + expert) : nullptr;
#pragma unroll
    for (int i = 0; i < 2; ++i)
#pragma unroll
        for (int j = 0; j < 4; ++j)
#pragma unroll
            for (int r = 0; r < 8; ++r) {
                const int ro = i * 16 + r;
                float* p = cbase + ro * LDC + j * 16;
                if (MODE == 1) {
                    float g = gbase[ro * NEXP];
                    *p += g * acc[i][j][r];
                } else {
                    *p = acc[i][j][r];
                }
            }
}

// ---------------- launcher ----------------
extern "C" void kernel_launch(void* const* d_in, const int* in_sizes, int n_in,
                              void* d_out, int out_size, void* d_ws, size_t ws_size,
                              hipStream_t stream) {
    const float* x         = (const float*)d_in[0];
    const float* gate_w    = (const float*)d_in[1];
    const float* base_w1   = (const float*)d_in[2];
    const float* spline_w1 = (const float*)d_in[3];
    const float* scaler_w1 = (const float*)d_in[4];
    const float* base_w2   = (const float*)d_in[5];
    const float* spline_w2 = (const float*)d_in[6];
    const float* scaler_w2 = (const float*)d_in[7];
    const float* base_w3   = (const float*)d_in[8];
    const float* spline_w3 = (const float*)d_in[9];
    const float* scaler_w3 = (const float*)d_in[10];
    float* out = (float*)d_out;

    char* ws = (char*)d_ws;
    size_t off = 0;
    auto carve = [&](size_t bytes) -> void* {
        void* p = ws + off;
        off += (bytes + 255) & ~(size_t)255;
        return p;
    };
    unsigned short* A1    = (unsigned short*)carve((size_t)NTOK * K1 * 2);
    unsigned short* A3    = (unsigned short*)carve((size_t)NTOK * K3 * 2);
    float*          H12   = (float*)carve((size_t)NTOK * N12 * 4);
    float*          gates = (float*)carve((size_t)NTOK * NEXP * 4);

    zero_f32<<<(NTOK * OUTD + 255) / 256, 256, 0, stream>>>(out, NTOK * OUTD);
    gate_kernel<<<NTOK, 32, 0, stream>>>(x, gate_w, gates);
    prep_a1<<<(NTOK * HID + 255) / 256, 256, 0, stream>>>(x, A1);

    for (int e = 0; e < NEXP; ++e) {
        const float* b1 = base_w1   + (size_t)e * DFF * HID;
        const float* s1 = spline_w1 + (size_t)e * DFF * HID * 8;
        const float* c1 = scaler_w1 + (size_t)e * DFF * HID;
        const float* b2 = base_w2   + (size_t)e * DFF * HID;
        const float* s2 = spline_w2 + (size_t)e * DFF * HID * 8;
        const float* c2 = scaler_w2 + (size_t)e * DFF * HID;
        const float* b3 = base_w3   + (size_t)e * OUTD * DFF;
        const float* s3 = spline_w3 + (size_t)e * OUTD * DFF * 8;
        const float* c3 = scaler_w3 + (size_t)e * OUTD * DFF;

        gemm_kan<0><<<dim3(N12 / 128, NTOK / 128), 256, 0, stream>>>(
            A1, b1, s1, c1, b2, s2, c2, H12, nullptr, 0);

        act_kernel<<<(NTOK * DFF + 255) / 256, 256, 0, stream>>>(H12, A3);

        gemm_kan<1><<<dim3(OUTD / 128, NTOK / 128), 256, 0, stream>>>(
            A3, b3, s3, c3, nullptr, nullptr, nullptr, out, gates, e);
    }
}
